// MHA_64141041598707
// MI455X (gfx1250) — matrix operand order repaired and checked by reading, not compiled, not run on hardware
//
#include <hip/hip_runtime.h>

// Problem constants (match reference)
#define BB 4
#define SS 2048
#define DD 256
#define HH 8
#define EE 32

typedef __attribute__((ext_vector_type(16))) _Float16 v16h;
typedef __attribute__((ext_vector_type(8)))  _Float16 v8h;
typedef __attribute__((ext_vector_type(8)))  float    v8f;

__device__ __forceinline__ v8f wmma_f16(v16h a, v16h b, v8f c) {
  // D = A(16x32 f16) * B(32x16 f16) + C(16x16 f32)
  return __builtin_amdgcn_wmma_f32_16x16x32_f16(
      /*neg_a=*/false, a, /*neg_b=*/false, b,
      /*c_mod=*/(short)0, c, /*reuse_a=*/false, /*reuse_b=*/false);
}

// A-fragment (16x32, f16). Lane l holds row (l&15).
// l<16 : K = k0+0..15;  l>=16 : K = k0+16..31 (contiguous)
// Works for both global and LDS source pointers (compiler picks b128 ld).
__device__ __forceinline__ v16h load_a(const _Float16* base, int ld, int k0, int lane) {
  const _Float16* p = base + (size_t)(lane & 15) * ld + k0 + ((lane & 16) ? 16 : 0);
  v8h lo = *(const v8h*)p;
  v8h hi = *(const v8h*)(p + 8);
  v16h r;
#pragma unroll
  for (int i = 0; i < 8; ++i) { r[i] = lo[i]; r[8 + i] = hi[i]; }
  return r;
}

// B-fragment (32x16, f16) from Bt stored [n][k] (row = output column).
// Lane l holds col (l&15); K = k0 + (l<16 ? 0..15 : 16..31), contiguous.
__device__ __forceinline__ v16h load_b(const _Float16* base, int ld, int k0, int lane) {
  const _Float16* p = base + (size_t)(lane & 15) * ld + k0 + ((lane & 16) ? 16 : 0);
  return *(const v16h*)p;
}

// Async copy 16B global -> LDS (per-lane), tracked by ASYNCcnt.
__device__ __forceinline__ void async_copy_b128(unsigned lds_off, const _Float16* g) {
  asm volatile("global_load_async_to_lds_b128 %0, %1, off"
               :: "v"(lds_off), "v"(g) : "memory");
}
__device__ __forceinline__ void wait_async_le1() {
  asm volatile("s_wait_asynccnt 0x1" ::: "memory");
}
__device__ __forceinline__ void wait_async_le0() {
  asm volatile("s_wait_asynccnt 0x0" ::: "memory");
}
// Generic LDS address truncated to 32 bits == DS byte offset (ISA 10.2).
__device__ __forceinline__ unsigned lds_offset(const void* p) {
  return (unsigned)(uintptr_t)p;
}

// ---------------- prep: fp32 -> fp16 conversions / weight transposes --------

__global__ void prep_x_kernel(const float* __restrict__ x, _Float16* __restrict__ xh, int n) {
  int i = blockIdx.x * blockDim.x + threadIdx.x;
  if (i < n) xh[i] = (_Float16)x[i];
}

__global__ void prep_w_kernel(const float* __restrict__ wq, const float* __restrict__ wk,
                              const float* __restrict__ wv, const float* __restrict__ wo,
                              _Float16* __restrict__ wqT, _Float16* __restrict__ wkT,
                              _Float16* __restrict__ wvT, _Float16* __restrict__ woT) {
  int i = blockIdx.x * blockDim.x + threadIdx.x;  // 0 .. H*D*E-1 (== D*D-1)
  // [h][d][e] -> [h][e][d]
  int h = i / (DD * EE);
  int rem = i - h * (DD * EE);
  int d = rem / EE;
  int e = rem - d * EE;
  int dst = h * (EE * DD) + e * DD + d;
  // fold softmax scale (D^-0.5) and log2(e) into Q weights -> exp2 softmax
  const float QSCALE = 0.0625f * 1.44269504088896340736f;
  wqT[dst] = (_Float16)(wq[i] * QSCALE);
  wkT[dst] = (_Float16)wk[i];
  wvT[dst] = (_Float16)wv[i];
  // wo [r][c] -> woT [c][r]
  int r = i / DD, c = i - r * DD;
  woT[c * DD + r] = (_Float16)wo[i];
}

// ---------------- QKV projection (WMMA) -------------------------------------
// One wave: 16 rows of s for one (b,h): q,k -> [B,H,S,E]; v -> vT [B,H,E,S]
__global__ void proj_kernel(const _Float16* __restrict__ xh,
                            const _Float16* __restrict__ wqT,
                            const _Float16* __restrict__ wkT,
                            const _Float16* __restrict__ wvT,
                            _Float16* __restrict__ qh,
                            _Float16* __restrict__ kh,
                            _Float16* __restrict__ vTh) {
  int lane = threadIdx.x & 31;
  int wave = threadIdx.x >> 5;
  int bh = blockIdx.y, b = bh >> 3, h = bh & 7;
  int s0 = blockIdx.x * 128 + wave * 16;

  const _Float16* xrow = xh + ((size_t)b * SS + s0) * DD;  // A for q/k ; B^T view for v
  const _Float16* wq = wqT + (size_t)h * EE * DD;
  const _Float16* wk = wkT + (size_t)h * EE * DD;
  const _Float16* wv = wvT + (size_t)h * EE * DD;

  v8f aq0 = 0, aq1 = 0, ak0 = 0, ak1 = 0, av0 = 0, av1 = 0;
#pragma unroll 2
  for (int kd = 0; kd < DD; kd += 32) {
    v16h ax  = load_a(xrow, DD, kd, lane);
    v16h bx  = load_b(xrow, DD, kd, lane);
    v16h bq0 = load_b(wq,           DD, kd, lane);
    v16h bq1 = load_b(wq + 16 * DD, DD, kd, lane);
    v16h bk0 = load_b(wk,           DD, kd, lane);
    v16h bk1 = load_b(wk + 16 * DD, DD, kd, lane);
    v16h aw0 = load_a(wv,           DD, kd, lane);
    v16h aw1 = load_a(wv + 16 * DD, DD, kd, lane);
    aq0 = wmma_f16(ax, bq0, aq0);
    aq1 = wmma_f16(ax, bq1, aq1);
    ak0 = wmma_f16(ax, bk0, ak0);
    ak1 = wmma_f16(ax, bk1, ak1);
    av0 = wmma_f16(aw0, bx, av0);   // vT tile: (M=e, N=s)
    av1 = wmma_f16(aw1, bx, av1);
  }

  // q,k C layout: VGPR r -> row s0 + r + (lane&16?8:0), col e = lane&15 (+16 tile1)
  int rbase = s0 + ((lane & 16) ? 8 : 0);
  int e = lane & 15;
  _Float16* qdst = qh + (size_t)bh * SS * EE;
  _Float16* kdst = kh + (size_t)bh * SS * EE;
#pragma unroll
  for (int r = 0; r < 8; ++r) {
    size_t row = (size_t)(rbase + r);
    qdst[row * EE + e]      = (_Float16)aq0[r];
    qdst[row * EE + 16 + e] = (_Float16)aq1[r];
    kdst[row * EE + e]      = (_Float16)ak0[r];
    kdst[row * EE + 16 + e] = (_Float16)ak1[r];
  }
  // vT C layout: VGPR r -> row e = r + (lane&16?8:0) (+16 tile1), col s = s0+(lane&15)
  _Float16* vdst = vTh + (size_t)bh * EE * SS;
  int scol = s0 + (lane & 15);
  int ebase = (lane & 16) ? 8 : 0;
#pragma unroll
  for (int r = 0; r < 8; ++r) {
    vdst[(size_t)(ebase + r) * SS + scol]      = (_Float16)av0[r];
    vdst[(size_t)(16 + ebase + r) * SS + scol] = (_Float16)av1[r];
  }
}

// ---------------- flash attention (transposed tiles, WMMA, LDS-staged KV) ---
// Workgroup: 8 waves x 16 query rows, one (b,h). KV tiles of 32 are staged
// into LDS once per workgroup via async copies (double buffered), so the
// 8 waves share each K/V byte instead of re-fetching it from cache.
__global__ void __launch_bounds__(256)
attn_kernel(const _Float16* __restrict__ qh,
            const _Float16* __restrict__ kh,
            const _Float16* __restrict__ vTh,
            _Float16* __restrict__ ch) {
  __shared__ _Float16 kbuf[2][32 * EE];  // [j_local][e], ld = 32
  __shared__ _Float16 vbuf[2][EE * 32];  // [e][j_local], ld = 32

  int tid = threadIdx.x;
  int lane = tid & 31;
  int wave = tid >> 5;
  int bh = blockIdx.y, b = bh >> 3, h = bh & 7;
  int i0 = blockIdx.x * 128 + wave * 16;

  const _Float16* qhead = qh + (size_t)bh * SS * EE;
  const _Float16* khead = kh + (size_t)bh * SS * EE;
  const _Float16* vhead = vTh + (size_t)bh * EE * SS;

  // Q as B-fragment (K=e is the full 32-wide contraction; loaded once)
  v16h bq = load_b(qhead + (size_t)i0 * EE, EE, 0, lane);

  v8f acc0 = 0, acc1 = 0;           // ctx^T tiles: (M=e 0..15 / 16..31, N=i)
  float m = -3.0e38f, l = 0.0f;     // per-lane: row i = i0 + (lane&15)
  v8f zc = 0;
  bool lo_half = (lane & 16) == 0;

  // cooperative async fill of one KV tile: 256 threads x 16B = 4KB
  // K half (waves 0-3): row j = t>>1;  V half (waves 4-7): row e = t>>1
  int fr = (tid & 127) >> 1;            // row within tile
  int fo = (tid & 1) * 8;               // half-offset in _Float16 elements
  bool fill_k = tid < 128;

  auto fill = [&](int buf, int j0) {
    if (fill_k) {
      async_copy_b128(lds_offset(&kbuf[buf][fr * 32 + fo]),
                      khead + (size_t)(j0 + fr) * EE + fo);
    } else {
      async_copy_b128(lds_offset(&vbuf[buf][fr * 32 + fo]),
                      vhead + (size_t)fr * SS + j0 + fo);
    }
  };

  const int NT = SS / 32;
  fill(0, 0);

  for (int jt = 0; jt < NT; ++jt) {
    int cur = jt & 1;
    if (jt + 1 < NT) {
      fill(cur ^ 1, (jt + 1) * 32);
      wait_async_le1();   // retire current tile's copy (in-order completion)
    } else {
      wait_async_le0();
    }
    __syncthreads();      // all waves' copies for `cur` visible

    // scores^T tiles: (M=j local, N=i), K/Q f16 from LDS/regs
    v16h akA = load_a(&kbuf[cur][0],       32, 0, lane);
    v16h akB = load_a(&kbuf[cur][16 * 32], 32, 0, lane);
    v8f s0 = wmma_f16(akA, bq, zc);
    v8f s1 = wmma_f16(akB, bq, zc);

    // online softmax: per-lane max over half the j's, one half-swap to finish
    float vm = s0[0];
#pragma unroll
    for (int r = 0; r < 8; ++r) { vm = fmaxf(vm, s0[r]); vm = fmaxf(vm, s1[r]); }
    vm = fmaxf(vm, __shfl_xor(vm, 16, 32));
    float mn = fmaxf(m, vm);
    float alpha = __builtin_amdgcn_exp2f(m - mn);
    m = mn;

    v8f p0, p1;
    float rs = 0.0f;
#pragma unroll
    for (int r = 0; r < 8; ++r) {
      p0[r] = __builtin_amdgcn_exp2f(s0[r] - mn);
      p1[r] = __builtin_amdgcn_exp2f(s1[r] - mn);
      rs += p0[r] + p1[r];
    }
    rs += __shfl_xor(rs, 16, 32);
    l = l * alpha + rs;
#pragma unroll
    for (int r = 0; r < 8; ++r) { acc0[r] *= alpha; acc1[r] *= alpha; }

    // P^T (C layout) -> B-fragment (K=j 0..31, N=i): half-wave swap + pack
    v16h pB;
#pragma unroll
    for (int r = 0; r < 8; ++r) {
      float x0 = __shfl_xor(p0[r], 16, 32);
      float x1 = __shfl_xor(p1[r], 16, 32);
      pB[r]     = (_Float16)(lo_half ? p0[r] : x1);
      pB[8 + r] = (_Float16)(lo_half ? x0 : p1[r]);
    }

    // ctx^T += V^T(16x32 over j) x P^T, V f16 from LDS
    v16h av0 = load_a(&vbuf[cur][0],       32, 0, lane);
    v16h av1 = load_a(&vbuf[cur][16 * 32], 32, 0, lane);
    acc0 = wmma_f16(av0, pB, acc0);
    acc1 = wmma_f16(av1, pB, acc1);

    __syncthreads();      // everyone done reading `cur` before it is refilled
  }

  float inv = 1.0f / l;
  // store into head-concat layout ch[b, s, h*E + e]; e runs along VGPR index
  _Float16* crow = ch + ((size_t)b * SS + i0 + (lane & 15)) * DD + h * EE + ((lane & 16) ? 8 : 0);
  v8h o0, o1;
#pragma unroll
  for (int r = 0; r < 8; ++r) {
    o0[r] = (_Float16)(acc0[r] * inv);
    o1[r] = (_Float16)(acc1[r] * inv);
  }
  *(v8h*)(crow)      = o0;   // e = base..base+7
  *(v8h*)(crow + 16) = o1;   // e = 16+base..
}

// ---------------- output projection (WMMA) ----------------------------------
// One wave: 16 rows (flattened b*s) x 64 output columns
__global__ void oproj_kernel(const _Float16* __restrict__ ch,
                             const _Float16* __restrict__ woT,
                             float* __restrict__ out) {
  int lane = threadIdx.x & 31;
  int wave = threadIdx.x >> 5;
  int s0 = blockIdx.x * 128 + wave * 16;  // over B*S
  int j0 = blockIdx.y * 64;

  const _Float16* arow = ch + (size_t)s0 * DD;
  v8f acc[4];
#pragma unroll
  for (int t = 0; t < 4; ++t) acc[t] = 0;

#pragma unroll 2
  for (int kd = 0; kd < DD; kd += 32) {
    v16h a = load_a(arow, DD, kd, lane);
#pragma unroll
    for (int t = 0; t < 4; ++t) {
      v16h bw = load_b(woT + (size_t)(j0 + 16 * t) * DD, DD, kd, lane);
      acc[t] = wmma_f16(a, bw, acc[t]);
    }
  }
  int rbase = s0 + ((lane & 16) ? 8 : 0);
  int jcol = lane & 15;
#pragma unroll
  for (int t = 0; t < 4; ++t)
#pragma unroll
    for (int r = 0; r < 8; ++r)
      out[(size_t)(rbase + r) * DD + j0 + 16 * t + jcol] = acc[t][r];
}

// ---------------- launch ----------------------------------------------------

extern "C" void kernel_launch(void* const* d_in, const int* in_sizes, int n_in,
                              void* d_out, int out_size, void* d_ws, size_t ws_size,
                              hipStream_t stream) {
  (void)in_sizes; (void)n_in; (void)out_size; (void)ws_size;
  const float* x  = (const float*)d_in[0];
  const float* wq = (const float*)d_in[1];
  const float* wk = (const float*)d_in[2];
  const float* wv = (const float*)d_in[3];
  const float* wo = (const float*)d_in[4];
  float* out = (float*)d_out;

  char* ws = (char*)d_ws;
  size_t off = 0;
  _Float16* xh  = (_Float16*)(ws + off); off += (size_t)BB * SS * DD * 2;       // 4 MiB
  _Float16* qh  = (_Float16*)(ws + off); off += (size_t)BB * HH * SS * EE * 2;  // 4 MiB
  _Float16* kh  = (_Float16*)(ws + off); off += (size_t)BB * HH * SS * EE * 2;  // 4 MiB
  _Float16* vTh = (_Float16*)(ws + off); off += (size_t)BB * HH * EE * SS * 2;  // 4 MiB
  _Float16* ch  = (_Float16*)(ws + off); off += (size_t)BB * SS * DD * 2;       // 4 MiB
  _Float16* wqT = (_Float16*)(ws + off); off += (size_t)HH * EE * DD * 2;       // 128 KiB
  _Float16* wkT = (_Float16*)(ws + off); off += (size_t)HH * EE * DD * 2;
  _Float16* wvT = (_Float16*)(ws + off); off += (size_t)HH * EE * DD * 2;
  _Float16* woT = (_Float16*)(ws + off); off += (size_t)DD * DD * 2;

  prep_x_kernel<<<(BB * SS * DD) / 256, 256, 0, stream>>>(x, xh, BB * SS * DD);
  prep_w_kernel<<<(HH * DD * EE) / 256, 256, 0, stream>>>(wq, wk, wv, wo, wqT, wkT, wvT, woT);

  dim3 g1(SS / 128, BB * HH);
  proj_kernel<<<g1, 256, 0, stream>>>(xh, wqT, wkT, wvT, qh, kh, vTh);
  attn_kernel<<<g1, 256, 0, stream>>>(qh, kh, vTh, ch);

  dim3 g2((BB * SS) / 128, DD / 64);
  oproj_kernel<<<g2, 256, 0, stream>>>(ch, woT, out);
}
